// CausalRCNBlock_46858093199882
// MI455X (gfx1250) — matrix operand order repaired and checked by reading, not compiled, not run on hardware
//
#include <hip/hip_runtime.h>
#include <hip/hip_bf16.h>
#include <math.h>

// ---------------- problem constants (match reference) ----------------
#define BDIM   8
#define LDIM   4096
#define HDIM   1024
#define KW     4
#define MROWS  (BDIM * LDIM)          // 32768 rows for every GEMM
#define N_SRU  (3 * HDIM)             // 3072
#define N_FF1  (2 * HDIM)             // 2048

typedef __attribute__((ext_vector_type(16))) __bf16 v16bf;
typedef __attribute__((ext_vector_type(8)))  float  v8f;

union FragU {
  uint4        q[2];
  unsigned int u[8];
  v16bf        v;
};

// ---------------- scalar bf16 helpers (RNE) ----------------
__device__ __forceinline__ __bf16 f2bf(float f) {
  unsigned int u = __float_as_uint(f);
  unsigned int r = u + 0x7FFFu + ((u >> 16) & 1u);
  unsigned short s = (unsigned short)(r >> 16);
  __bf16 b;
  __builtin_memcpy(&b, &s, 2);
  return b;
}
__device__ __forceinline__ float bf2f(__bf16 b) {
  unsigned short s;
  __builtin_memcpy(&s, &b, 2);
  return __uint_as_float(((unsigned int)s) << 16);
}

// ---------------- WMMA fragment loaders ----------------
// A (16x32 bf16, M x K): lane l owns row m=l&15; K-half kh=(l>>4)*8.
// VGPR0..3 = K kh..kh+7, VGPR4..7 = K kh+16..kh+23  -> two contiguous 16B loads.
__device__ __forceinline__ v16bf load_a_row(const __bf16* rowp, int kbase, int lane) {
  const int kh = (lane >> 4) << 3;
  const __bf16* p = rowp + kbase + kh;
  FragU r;
  r.q[0] = *reinterpret_cast<const uint4*>(p);
  r.q[1] = *reinterpret_cast<const uint4*>(p + 16);
  return r.v;
}
// B (32x16 bf16, K x N), weights stored COLUMN-major: Bcol[n][Kd].
// lane l owns col n=l&15; kh=(l>>4)*8; VGPR0..3 = K kh..kh+7, VGPR4..7 = K kh+16..kh+23
// -> two contiguous 16B loads.
__device__ __forceinline__ v16bf load_b_col(const __bf16* colbase, size_t ldb,
                                            int kbase, int lane) {
  const int n  = lane & 15;
  const int kh = (lane >> 4) << 3;
  const __bf16* p = colbase + (size_t)n * ldb + kbase + kh;
  FragU r;
  r.q[0] = *reinterpret_cast<const uint4*>(p);
  r.q[1] = *reinterpret_cast<const uint4*>(p + 16);
  return r.v;
}
__device__ __forceinline__ v16bf zero_frag() {
  FragU r;
#pragma unroll
  for (int i = 0; i < 8; ++i) r.u[i] = 0u;
  return r.v;
}

// ---------------- generic GEMM: C[M,N] = A[M,Kd](bf16) x Bcol[N,Kd](bf16) ----------------
// wave tile = 16(M) x 64(N): 1 A-frag feeds 4 WMMAs per 32-K chunk.
__global__ void __launch_bounds__(256)
gemm_bf16_wmma(const __bf16* __restrict__ A, const __bf16* __restrict__ Bcol,
               float* __restrict__ C, const float* __restrict__ bias,
               const float* __restrict__ resid, int Kd, int N) {
  const int lane    = threadIdx.x & 31;
  const int gw      = blockIdx.x * 8 + (threadIdx.x >> 5);
  const int stripsN = N >> 6;
  const int tileM   = (gw / stripsN) << 4;
  const int tileN   = (gw % stripsN) << 6;

  const __bf16* rowp = A + ((size_t)tileM + (lane & 15)) * (size_t)Kd;
  const __bf16* c0 = Bcol + (size_t)(tileN +  0) * Kd;
  const __bf16* c1 = Bcol + (size_t)(tileN + 16) * Kd;
  const __bf16* c2 = Bcol + (size_t)(tileN + 32) * Kd;
  const __bf16* c3 = Bcol + (size_t)(tileN + 48) * Kd;

  v8f a0 = {}, a1 = {}, a2 = {}, a3 = {};
  for (int kb = 0; kb < Kd; kb += 32) {
    __builtin_prefetch(rowp + kb + 256, 0, 1);   // global_prefetch_b8 on the A stream
    v16bf af = load_a_row(rowp, kb, lane);
    v16bf b0 = load_b_col(c0, Kd, kb, lane);
    v16bf b1 = load_b_col(c1, Kd, kb, lane);
    v16bf b2 = load_b_col(c2, Kd, kb, lane);
    v16bf b3 = load_b_col(c3, Kd, kb, lane);
    a0 = __builtin_amdgcn_wmma_f32_16x16x32_bf16(false, af, false, b0, (short)0, a0, false, false);
    a1 = __builtin_amdgcn_wmma_f32_16x16x32_bf16(false, af, false, b1, (short)0, a1, false, false);
    a2 = __builtin_amdgcn_wmma_f32_16x16x32_bf16(false, af, false, b2, (short)0, a2, false, false);
    a3 = __builtin_amdgcn_wmma_f32_16x16x32_bf16(false, af, false, b3, (short)0, a3, false, false);
  }

  // C/D layout: lanes 0-15 -> N=lane, rows 0-7 in VGPR v; lanes 16-31 -> rows 8-15.
  const int n0 = tileN + (lane & 15);
  const int m0 = tileM + ((lane >> 4) << 3);
#pragma unroll
  for (int v = 0; v < 8; ++v) {
    const size_t r = (size_t)(m0 + v) * (size_t)N;
    float o0 = a0[v], o1 = a1[v], o2 = a2[v], o3 = a3[v];
    if (bias)  { o0 += bias[n0];      o1 += bias[n0 + 16];
                 o2 += bias[n0 + 32]; o3 += bias[n0 + 48]; }
    if (resid) { o0 += resid[r + n0];      o1 += resid[r + n0 + 16];
                 o2 += resid[r + n0 + 32]; o3 += resid[r + n0 + 48]; }
    C[r + n0]      = o0; C[r + n0 + 16] = o1;
    C[r + n0 + 32] = o2; C[r + n0 + 48] = o3;
  }
}

// ---------------- causal conv as 4 shifted-A GEMM accumulations ----------------
__global__ void __launch_bounds__(256)
conv_gemm_wmma(const __bf16* __restrict__ xbf, const __bf16* __restrict__ wconv,
               const float* __restrict__ conv_b, float* __restrict__ cnn) {
  const int lane    = threadIdx.x & 31;
  const int gw      = blockIdx.x * 8 + (threadIdx.x >> 5);
  const int stripsN = HDIM >> 6;                 // 16
  const int tileM   = (gw / stripsN) << 4;
  const int tileN   = (gw % stripsN) << 6;
  const int bIdx    = tileM / LDIM;
  const int t0      = tileM % LDIM;

  v8f a0 = {}, a1 = {}, a2 = {}, a3 = {};
  const __bf16* xb = xbf + (size_t)bIdx * LDIM * HDIM;

  for (int k = 0; k < KW; ++k) {
    const int  tr    = t0 + (lane & 15) + k - (KW - 1);   // shifted time row this lane owns
    const bool valid = (tr >= 0);
    const __bf16* rowp = xb + (size_t)(valid ? tr : 0) * HDIM;
    const __bf16* wk   = wconv + (size_t)k * HDIM * HDIM;
    const __bf16* c0 = wk + (size_t)(tileN +  0) * HDIM;
    const __bf16* c1 = wk + (size_t)(tileN + 16) * HDIM;
    const __bf16* c2 = wk + (size_t)(tileN + 32) * HDIM;
    const __bf16* c3 = wk + (size_t)(tileN + 48) * HDIM;
    for (int kb = 0; kb < HDIM; kb += 32) {
      v16bf af = valid ? load_a_row(rowp, kb, lane) : zero_frag();
      v16bf b0 = load_b_col(c0, HDIM, kb, lane);
      v16bf b1 = load_b_col(c1, HDIM, kb, lane);
      v16bf b2 = load_b_col(c2, HDIM, kb, lane);
      v16bf b3 = load_b_col(c3, HDIM, kb, lane);
      a0 = __builtin_amdgcn_wmma_f32_16x16x32_bf16(false, af, false, b0, (short)0, a0, false, false);
      a1 = __builtin_amdgcn_wmma_f32_16x16x32_bf16(false, af, false, b1, (short)0, a1, false, false);
      a2 = __builtin_amdgcn_wmma_f32_16x16x32_bf16(false, af, false, b2, (short)0, a2, false, false);
      a3 = __builtin_amdgcn_wmma_f32_16x16x32_bf16(false, af, false, b3, (short)0, a3, false, false);
    }
  }

  const int n0 = tileN + (lane & 15);
  const int m0 = tileM + ((lane >> 4) << 3);
#pragma unroll
  for (int v = 0; v < 8; ++v) {
    const size_t r = (size_t)(m0 + v) * HDIM;
    cnn[r + n0]      = a0[v] + conv_b[n0];
    cnn[r + n0 + 16] = a1[v] + conv_b[n0 + 16];
    cnn[r + n0 + 32] = a2[v] + conv_b[n0 + 32];
    cnn[r + n0 + 48] = a3[v] + conv_b[n0 + 48];
  }
}

// ---------------- block reduction helper ----------------
__device__ __forceinline__ float block_reduce_sum(float v, float* red, int tid) {
  red[tid] = v;
  __syncthreads();
  for (int s = 128; s > 0; s >>= 1) {
    if (tid < s) red[tid] += red[tid + s];
    __syncthreads();
  }
  float r = red[0];
  __syncthreads();
  return r;
}

__device__ __forceinline__ float gelu_exact(float y) {
  return 0.5f * y * (1.0f + erff(y * 0.70710678118654752f));
}

// ---------------- row LayerNorm -> bf16 (optional exact gelu) ----------------
__global__ void __launch_bounds__(256)
ln_rows_bf16(const float* __restrict__ in, const float* __restrict__ w,
             const float* __restrict__ bsh, __bf16* __restrict__ out,
             int width, int do_gelu) {
  __shared__ float red[256];
  const int row = blockIdx.x;
  const int tid = threadIdx.x;
  const int epc = width >> 8;                    // 4 (H) or 8 (2H)
  const float* rp = in + (size_t)row * width;

  float vals[8];
  float s = 0.0f;
  for (int i = 0; i < epc; ++i) { float v = rp[tid + (i << 8)]; vals[i] = v; s += v; }
  const float mean = block_reduce_sum(s, red, tid) / (float)width;
  float q = 0.0f;
  for (int i = 0; i < epc; ++i) { float d = vals[i] - mean; q += d * d; }
  const float rs = rsqrtf(block_reduce_sum(q, red, tid) / (float)width + 1e-5f);
  for (int i = 0; i < epc; ++i) {
    const int c = tid + (i << 8);
    float y = (vals[i] - mean) * rs * w[c] + bsh[c];
    if (do_gelu) y = gelu_exact(y);
    out[(size_t)row * width + c] = f2bf(y);
  }
}

// ---------------- SRU scan + lambda blend (in-place over cnn) ----------------
__global__ void __launch_bounds__(256)
sru_scan_blend(const float* __restrict__ U, const __bf16* __restrict__ xn,
               float* __restrict__ cnn,  // in: cnn, out: blended pre-RMS
               const float* __restrict__ vf, const float* __restrict__ vr,
               const float* __restrict__ bf_, const float* __restrict__ br,
               const float* __restrict__ lam, float scale_x) {
  const int gid = blockIdx.x * 256 + threadIdx.x;   // 0 .. B*H-1
  const int b   = gid >> 10;
  const int h   = gid & (HDIM - 1);
  const float vfh = vf[h], vrh = vr[h], bfh = bf_[h], brh = br[h], lh = lam[h];
  float c = 0.0f;
  const size_t base = (size_t)b * LDIM;
  for (int t = 0; t < LDIM; ++t) {
    const size_t m = base + t;
    const float* up = U + m * (size_t)N_SRU;
    const float u0 = up[h], u1 = up[HDIM + h], u2 = up[2 * HDIM + h];
    const float xr = bf2f(xn[m * HDIM + h]) * scale_x;
    const float f  = 1.0f / (1.0f + __expf(-(u1 + c * vfh + bfh)));
    const float cn = f * c + (1.0f - f) * u0;
    const float r  = 1.0f / (1.0f + __expf(-(u2 + c * vrh + brh)));
    const float hv = r * tanhf(cn) + (1.0f - r) * xr;
    c = cn;
    const float cv = cnn[m * HDIM + h];
    cnn[m * HDIM + h] = lh * cv + (1.0f - lh) * hv;
  }
}

// ---------------- RMSNorm * x gate (in place) + LN1 + gelu -> bf16 ----------------
__global__ void __launch_bounds__(256)
rms_gate_ln1(float* __restrict__ buf /* in: blended, out: gated */,
             const float* __restrict__ x, const float* __restrict__ rms_w,
             const float* __restrict__ ln1_w, const float* __restrict__ ln1_b,
             __bf16* __restrict__ t1) {
  __shared__ float red[256];
  const int row = blockIdx.x;
  const int tid = threadIdx.x;
  float* rp = buf + (size_t)row * HDIM;
  const float* xp = x + (size_t)row * HDIM;

  float v[4], g[4];
  float q = 0.0f;
  for (int i = 0; i < 4; ++i) { v[i] = rp[tid + (i << 8)]; q += v[i] * v[i]; }
  const float scale = rsqrtf(block_reduce_sum(q, red, tid) / (float)HDIM + 1e-6f);

  float s = 0.0f;
  for (int i = 0; i < 4; ++i) {
    const int c = tid + (i << 8);
    g[i] = rms_w[c] * v[i] * scale * xp[c];
    rp[c] = g[i];                                   // gated (FFN residual)
    s += g[i];
  }
  const float mean = block_reduce_sum(s, red, tid) / (float)HDIM;
  float q2 = 0.0f;
  for (int i = 0; i < 4; ++i) { float d = g[i] - mean; q2 += d * d; }
  const float rs = rsqrtf(block_reduce_sum(q2, red, tid) / (float)HDIM + 1e-5f);
  for (int i = 0; i < 4; ++i) {
    const int c = tid + (i << 8);
    const float y = (g[i] - mean) * rs * ln1_w[c] + ln1_b[c];
    t1[(size_t)row * HDIM + c] = f2bf(gelu_exact(y));
  }
}

// ---------------- conversion / packing ----------------
__global__ void __launch_bounds__(256)
f32_to_bf16(const float* __restrict__ in, __bf16* __restrict__ out, int n) {
  const int i = blockIdx.x * 256 + threadIdx.x;
  if (i < n) out[i] = f2bf(in[i]);
}
// row-major [Kd][N] fp32 -> column-major [N][Kd] bf16
__global__ void __launch_bounds__(256)
pack_transpose_bf16(const float* __restrict__ in, __bf16* __restrict__ out,
                    int Kd, int N) {
  const int idx = blockIdx.x * 256 + threadIdx.x;
  if (idx < Kd * N) {
    const int i = idx / N, n = idx % N;
    out[(size_t)n * Kd + i] = f2bf(in[idx]);
  }
}
// conv_w (O, I, K) fp32 -> per-k column-major [k][O][I] bf16
__global__ void __launch_bounds__(256)
pack_convw(const float* __restrict__ in, __bf16* __restrict__ out) {
  const int idx = blockIdx.x * 256 + threadIdx.x;   // over H*H*KW outputs
  if (idx < HDIM * HDIM * KW) {
    const int k = idx / (HDIM * HDIM);
    const int r = idx % (HDIM * HDIM);
    const int o = r / HDIM, i = r % HDIM;
    out[idx] = f2bf(in[(size_t)o * HDIM * KW + (size_t)i * KW + k]);
  }
}

// ---------------- host-side launch ----------------
extern "C" void kernel_launch(void* const* d_in, const int* in_sizes, int n_in,
                              void* d_out, int out_size, void* d_ws, size_t ws_size,
                              hipStream_t stream) {
  const float* x        = (const float*)d_in[0];
  const float* conv_w   = (const float*)d_in[1];
  const float* conv_b   = (const float*)d_in[2];
  const float* sru_ln_w = (const float*)d_in[3];
  const float* sru_ln_b = (const float*)d_in[4];
  const float* sru_W    = (const float*)d_in[5];
  const float* sru_vf   = (const float*)d_in[6];
  const float* sru_vr   = (const float*)d_in[7];
  const float* sru_bf   = (const float*)d_in[8];
  const float* sru_br   = (const float*)d_in[9];
  const float* lambda_w = (const float*)d_in[10];
  const float* rms_w    = (const float*)d_in[11];
  const float* ln1_w    = (const float*)d_in[12];
  const float* ln1_b    = (const float*)d_in[13];
  const float* lin1_w   = (const float*)d_in[14];
  const float* ln2_w    = (const float*)d_in[15];
  const float* ln2_b    = (const float*)d_in[16];
  const float* lin2_w   = (const float*)d_in[17];
  const float* lin2_b   = (const float*)d_in[18];
  float* out = (float*)d_out;

  // workspace layout (with lifetime-based aliasing, total 0x2D600000 ~ 726 MiB)
  char* ws = (char*)d_ws;
  __bf16* xbf   = (__bf16*)(ws + 0x00000000);  // 64 MiB  (dead after conv)
  __bf16* xn    = (__bf16*)(ws + 0x04000000);  // 64 MiB  (dead after scan)
  __bf16* t2bf  = (__bf16*)(ws + 0x00000000);  // 128 MiB alias over xbf+xn
  float*  cnn   = (float*) (ws + 0x08000000);  // 128 MiB (cnn -> blended -> gated, in place)
  float*  U     = (float*) (ws + 0x10000000);  // 384 MiB (dead after scan)
  float*  ffn1  = (float*) (ws + 0x10000000);  // 256 MiB alias over U
  __bf16* t1bf  = (__bf16*)(ws + 0x28000000);  // 64 MiB
  __bf16* wconv = (__bf16*)(ws + 0x2C000000);  // 8 MiB
  __bf16* wsru  = (__bf16*)(ws + 0x2C800000);  // 6 MiB
  __bf16* wlin1 = (__bf16*)(ws + 0x2CE00000);  // 4 MiB
  __bf16* wlin2 = (__bf16*)(ws + 0x2D200000);  // 4 MiB

  const int totalX = MROWS * HDIM;                       // 33,554,432
  f32_to_bf16<<<totalX / 256, 256, 0, stream>>>(x, xbf, totalX);
  pack_convw<<<(HDIM * HDIM * KW) / 256, 256, 0, stream>>>(conv_w, wconv);
  pack_transpose_bf16<<<(HDIM * N_SRU) / 256, 256, 0, stream>>>(sru_W, wsru, HDIM, N_SRU);
  pack_transpose_bf16<<<(HDIM * N_FF1) / 256, 256, 0, stream>>>(lin1_w, wlin1, HDIM, N_FF1);
  pack_transpose_bf16<<<(N_FF1 * HDIM) / 256, 256, 0, stream>>>(lin2_w, wlin2, N_FF1, HDIM);

  // 1) causal conv  (32768x1024, K-depth 4096)
  conv_gemm_wmma<<<((MROWS / 16) * (HDIM / 64)) / 8, 256, 0, stream>>>(xbf, wconv, conv_b, cnn);
  // 2) SRU LayerNorm -> xn (bf16)
  ln_rows_bf16<<<MROWS, 256, 0, stream>>>(cnn, sru_ln_w, sru_ln_b, xn, HDIM, 0);
  // 3) U = xn @ sru_W  (32768x3072, K=1024)
  gemm_bf16_wmma<<<((MROWS / 16) * (N_SRU / 64)) / 8, 256, 0, stream>>>(
      xn, wsru, U, nullptr, nullptr, HDIM, N_SRU);
  // 4) SRU scan + lambda blend (in place over cnn)
  const float scale_x = sqrtf(1.0f + expf(2.0f * -2.0f));
  sru_scan_blend<<<(BDIM * HDIM) / 256, 256, 0, stream>>>(
      U, xn, cnn, sru_vf, sru_vr, sru_bf, sru_br, lambda_w, scale_x);
  // 5) RMSNorm * x gate (gated, in place) + LN1 + gelu -> t1
  rms_gate_ln1<<<MROWS, 256, 0, stream>>>(cnn, x, rms_w, ln1_w, ln1_b, t1bf);
  // 6) ffn1 = t1 @ lin1  (32768x2048, K=1024)
  gemm_bf16_wmma<<<((MROWS / 16) * (N_FF1 / 64)) / 8, 256, 0, stream>>>(
      t1bf, wlin1, ffn1, nullptr, nullptr, HDIM, N_FF1);
  // 7) LN2 + gelu -> t2 (bf16)
  ln_rows_bf16<<<MROWS, 256, 0, stream>>>(ffn1, ln2_w, ln2_b, t2bf, N_FF1, 1);
  // 8) out = t2 @ lin2 + lin2_b + gated  (32768x1024, K=2048)
  gemm_bf16_wmma<<<((MROWS / 16) * (HDIM / 64)) / 8, 256, 0, stream>>>(
      t2bf, wlin2, out, lin2_b, cnn /* gated residual */, N_FF1, HDIM);
}
